// RNN_Reverser_14070312861726
// MI455X (gfx1250) — matrix operand order repaired and checked by reading, not compiled, not run on hardware
//
#include <hip/hip_runtime.h>
#include <math.h>

// ---------------------------------------------------------------------------
// Seq2seq Elman RNN (2 layers, H=D=1024, B=128, 512 enc + 512 dec steps)
// Persistent grid-sync kernel, bf16 WMMA (v_wmma_f32_16x16x32_bf16), f32 acc.
// Layer-pipelined (1 grid barrier per step). A-operands staged to LDS with
// gfx1250 async global->LDS loads (ASYNCcnt), double-buffered. 2 m-tiles per
// wave for weight reuse; pipelined f32->bf16 input conversion.
// ---------------------------------------------------------------------------

#define TIN   513
#define TOUT  512
#define BATCH 128
#define DIM   1024          // D == H == 1024
#define NBLK  32
#define NTHR  256           // 8 wave32 per block
#define SROW  1040          // LDS row stride in bf16 (520 dwords = 8 mod 64 banks)
#define SELEM (32 * SROW)                 // elements per staging buffer
#define SBYTES (2 * SELEM * 2)            // two buffers -> 133120 B dynamic LDS

#if defined(__has_builtin)
#if __has_builtin(__builtin_amdgcn_global_load_async_to_lds_b128) && \
    __has_builtin(__builtin_amdgcn_s_wait_asynccnt)
#define USE_ASYNC_LDS 1
#endif
#endif

typedef __bf16 bf16_t;
typedef bf16_t v16bf __attribute__((ext_vector_type(16)));
typedef float  v8f   __attribute__((ext_vector_type(8)));

#ifdef USE_ASYNC_LDS
// builtin expects: (int4 addrspace(1)*, int4 addrspace(3)*, imm int, imm int)
typedef int v4i_g __attribute__((vector_size(16)));
typedef __attribute__((address_space(1))) v4i_g* gptr4;
typedef __attribute__((address_space(3))) v4i_g* lptr4;
#endif

static constexpr size_t WELEM   = (size_t)2 * DIM * DIM;   // one [L,H,D] tensor
static constexpr size_t LINELEM = (size_t)DIM * DIM;
static constexpr size_t HELEM   = (size_t)BATCH * DIM;

// workspace layout (bytes)
static constexpr size_t OFF_BAR  = 0;
static constexpr size_t OFF_EWIH = 256;                        // bf16 weights
static constexpr size_t OFF_EWHH = OFF_EWIH + WELEM * 2;
static constexpr size_t OFF_DWIH = OFF_EWHH + WELEM * 2;
static constexpr size_t OFF_DWHH = OFF_DWIH + WELEM * 2;
static constexpr size_t OFF_LINW = OFF_DWHH + WELEM * 2;
static constexpr size_t OFF_BENC = OFF_LINW + LINELEM * 2;     // f32 fused biases
static constexpr size_t OFF_BDEC = OFF_BENC + (size_t)2 * DIM * 4;
static constexpr size_t OFF_H0   = OFF_BDEC + (size_t)2 * DIM * 4; // bf16 x2 bufs
static constexpr size_t OFF_H1   = OFF_H0 + (size_t)2 * HELEM * 2; // bf16 x2 bufs
static constexpr size_t OFF_XBF  = OFF_H1 + (size_t)2 * HELEM * 2; // bf16 x2 bufs

// ---------------------------------------------------------------------------
// helpers
// ---------------------------------------------------------------------------
__device__ __forceinline__ unsigned f2bf_u(float f) {
    unsigned u = __float_as_uint(f);               // round-to-nearest-even
    return (u + 0x7FFFu + ((u >> 16) & 1u)) >> 16;
}
__device__ __forceinline__ unsigned short f2bf(float f) {
    return (unsigned short)f2bf_u(f);
}
__device__ __forceinline__ unsigned pack2(float lo, float hi) {
    return f2bf_u(lo) | (f2bf_u(hi) << 16);
}

union ABReg {
    v16bf    v;
    unsigned u[8];
    uint4    q[2];
};

__device__ __forceinline__ v8f wmma_bf16(const v16bf& a, const v16bf& b, v8f c) {
    return __builtin_amdgcn_wmma_f32_16x16x32_bf16(false, a, false, b, (short)0, c,
                                                   false, false);
}

// A-matrix 16x32 bf16 layout: lane (g=lane>>4, m=lane&15) holds row m;
// VGPR i<4 -> dword i+4g, VGPR i>=4 -> dword 8+(i-4)+4g of the 32-elem K chunk.
__device__ __forceinline__ void load_a_bf16(ABReg& r, const unsigned short* p16, int g) {
    const uint4* p = (const uint4*)p16;
    r.q[0] = p[g];
    r.q[1] = p[2 + g];
}
// B-matrix supplied as N-rows x K: lane n=lane&15 holds weight row n;
// same K layout as A: dwords i+4g (i<4), 8+(i-4)+4g (i>=4).
__device__ __forceinline__ void load_b_bf16(ABReg& r, const unsigned short* p16, int g) {
    const uint4* p = (const uint4*)p16;
    r.q[0] = p[g];
    r.q[1] = p[2 + g];
}

// Issue cooperative copy: 32 rows x 1024 bf16 global -> LDS (stride SROW).
// Async path: GLOBAL_LOAD_ASYNC_TO_LDS_B128, completion via ASYNCcnt.
__device__ __forceinline__ void stage_issue(unsigned short* sdst,
                                            const unsigned short* __restrict__ gsrc,
                                            int tid) {
#ifdef USE_ASYNC_LDS
    #pragma unroll
    for (int i = 0; i < 16; ++i) {
        const int idx = tid + i * NTHR;             // 0..4095
        const int row = idx >> 7, col = idx & 127;
        __builtin_amdgcn_global_load_async_to_lds_b128(
            (gptr4)(gsrc + row * DIM + col * 8),
            (lptr4)(sdst + row * SROW + col * 8),
            0, 0);
    }
#else
    const uint4* src = (const uint4*)gsrc;          // 128 uint4 per row
    #pragma unroll
    for (int h = 0; h < 2; ++h) {
        uint4 t[8];
        #pragma unroll
        for (int i = 0; i < 8; ++i) {
            const int idx = tid + (h * 8 + i) * NTHR;
            t[i] = src[(idx >> 7) * 128 + (idx & 127)];
        }
        #pragma unroll
        for (int i = 0; i < 8; ++i) {
            const int idx = tid + (h * 8 + i) * NTHR;
            *(uint4*)(sdst + (idx >> 7) * SROW + (idx & 127) * 8) = t[i];
        }
    }
#endif
}
// Wait for this wave's staged data, then make all waves' staging visible.
__device__ __forceinline__ void stage_wait() {
#ifdef USE_ASYNC_LDS
    __builtin_amdgcn_s_wait_asynccnt(0);
#endif
    __syncthreads();
}

// One GEMM pass: acc[mt] += A(LDS, 2 m-tiles) x W^T(ntile rows, global), K=DIM.
__device__ __forceinline__ void mm_accum(v8f acc[2], const unsigned short* sA,
                                         const unsigned short* __restrict__ wrow,
                                         int g, int l15) {
    #pragma unroll 4
    for (int kc = 0; kc < DIM; kc += 32) {
        ABReg b;
        load_b_bf16(b, wrow + kc, g);
        #pragma unroll
        for (int mt = 0; mt < 2; ++mt) {
            ABReg a;
            load_a_bf16(a, sA + (mt * 16 + l15) * SROW + kc, g);
            acc[mt] = wmma_bf16(a.v, b.v, acc[mt]);
        }
    }
}

// f32 [B,DIM] timestep -> bf16 staging buffer (whole grid cooperates)
__device__ __forceinline__ void convert_x(const float* __restrict__ src,
                                          unsigned short* __restrict__ dst,
                                          int tid, int nth) {
    const float4* s4 = (const float4*)src;
    uint2* d2 = (uint2*)dst;
    for (int i = tid; i < (int)(HELEM / 4); i += nth) {
        float4 f = s4[i];
        uint2 r; r.x = pack2(f.x, f.y); r.y = pack2(f.z, f.w);
        d2[i] = r;
    }
}

__device__ __forceinline__ void grid_barrier(unsigned* bar, unsigned& epoch) {
    __syncthreads();
    if (threadIdx.x == 0) {
        __threadfence();
        __hip_atomic_fetch_add(bar, 1u, __ATOMIC_RELEASE, __HIP_MEMORY_SCOPE_AGENT);
        epoch += NBLK;
        while (__hip_atomic_load(bar, __ATOMIC_ACQUIRE, __HIP_MEMORY_SCOPE_AGENT) < epoch) {
            __builtin_amdgcn_s_sleep(4);
        }
    }
    __syncthreads();
}

// ---------------------------------------------------------------------------
// kernels
// ---------------------------------------------------------------------------
__global__ void ws_init_kernel(unsigned* bar) {
    if (threadIdx.x == 0) *bar = 0u;
}

__global__ __launch_bounds__(NTHR) void rnn_persistent_kernel(
    const float* __restrict__ X,    const float* __restrict__ Y,
    const float* __restrict__ eWih, const float* __restrict__ eWhh,
    const float* __restrict__ eBih, const float* __restrict__ eBhh,
    const float* __restrict__ dWih, const float* __restrict__ dWhh,
    const float* __restrict__ dBih, const float* __restrict__ dBhh,
    const float* __restrict__ linW, const float* __restrict__ linB,
    float* __restrict__ out, char* __restrict__ ws)
{
    extern __shared__ unsigned short sLDS[];     // 2 staging buffers
    unsigned short* sA0 = sLDS;
    unsigned short* sA1 = sLDS + SELEM;

    unsigned*       bar   = (unsigned*)(ws + OFF_BAR);
    unsigned short* wEWih = (unsigned short*)(ws + OFF_EWIH);
    unsigned short* wEWhh = (unsigned short*)(ws + OFF_EWHH);
    unsigned short* wDWih = (unsigned short*)(ws + OFF_DWIH);
    unsigned short* wDWhh = (unsigned short*)(ws + OFF_DWHH);
    unsigned short* wLin  = (unsigned short*)(ws + OFF_LINW);
    float*          bEnc  = (float*)(ws + OFF_BENC);
    float*          bDec  = (float*)(ws + OFF_BDEC);
    unsigned short* h0buf[2] = {(unsigned short*)(ws + OFF_H0),
                                (unsigned short*)(ws + OFF_H0) + HELEM};
    unsigned short* h1buf[2] = {(unsigned short*)(ws + OFF_H1),
                                (unsigned short*)(ws + OFF_H1) + HELEM};
    unsigned short* xbf[2]   = {(unsigned short*)(ws + OFF_XBF),
                                (unsigned short*)(ws + OFF_XBF) + HELEM};

    const int tid = blockIdx.x * NTHR + threadIdx.x;
    const int nth = NBLK * NTHR;

    // ---- prep: bf16 weights (L2-resident), fused biases, zero hidden, x(0) ----
    {
        const float4* a4 = (const float4*)eWih;
        const float4* b4 = (const float4*)eWhh;
        const float4* c4 = (const float4*)dWih;
        const float4* d4 = (const float4*)dWhh;
        uint2* ad = (uint2*)wEWih; uint2* bd = (uint2*)wEWhh;
        uint2* cd = (uint2*)wDWih; uint2* dd = (uint2*)wDWhh;
        for (size_t i = tid; i < WELEM / 4; i += nth) {
            float4 f; uint2 r;
            f = a4[i]; r.x = pack2(f.x, f.y); r.y = pack2(f.z, f.w); ad[i] = r;
            f = b4[i]; r.x = pack2(f.x, f.y); r.y = pack2(f.z, f.w); bd[i] = r;
            f = c4[i]; r.x = pack2(f.x, f.y); r.y = pack2(f.z, f.w); cd[i] = r;
            f = d4[i]; r.x = pack2(f.x, f.y); r.y = pack2(f.z, f.w); dd[i] = r;
        }
        const float4* l4 = (const float4*)linW;
        uint2* ld = (uint2*)wLin;
        for (size_t i = tid; i < LINELEM / 4; i += nth) {
            float4 f = l4[i];
            uint2 r; r.x = pack2(f.x, f.y); r.y = pack2(f.z, f.w); ld[i] = r;
        }
        for (size_t i = tid; i < (size_t)2 * DIM; i += nth) {
            bEnc[i] = eBih[i] + eBhh[i];
            bDec[i] = dBih[i] + dBhh[i];
        }
        uint4 z; z.x = z.y = z.z = z.w = 0u;
        uint4* hz = (uint4*)h0buf[0];
        for (size_t i = tid; i < (size_t)4 * HELEM / 8; i += nth) hz[i] = z;
        convert_x(X, xbf[0], tid, nth);
    }

    unsigned epoch = 0;
    grid_barrier(bar, epoch);

    // tile mapping: 32 blocks x 8 waves; block owns 2 m-tiles (32 batch rows)
    // and 8 n-tiles (one per wave). Wave computes a 2x1 tile block.
    const int wave  = threadIdx.x >> 5;
    const int lane  = threadIdx.x & 31;
    const int g     = lane >> 4;
    const int l15   = lane & 15;
    const int mpair = blockIdx.x >> 3;                 // 0..3
    const int bbase = mpair * 32;                      // first batch row of block
    const int ntile = (blockIdx.x & 7) * 8 + wave;     // 0..63
    const size_t nroff = (size_t)(ntile * 16 + l15) * DIM;   // weight-row offset
    const int    ncol  = ntile * 16 + l15;

    // ---- pipelined recurrence: phase p computes h0(p), h1(p-1), logits(p-2)
    //      and converts x(p+1). One grid barrier per phase. ----
    for (int p = 0; p <= TOUT * 2 + 1; ++p) {          // p = 0..1025
        // convert next timestep's input (read next phase)
        if (p + 1 <= 2 * TOUT - 1) {
            const int s = p + 1;
            const float* xsrc =
                (s < TOUT)  ? X + (size_t)s * HELEM :
                (s == TOUT) ? X + (size_t)TOUT * HELEM :
                              Y + (size_t)(s - TOUT - 1) * HELEM;
            convert_x(xsrc, xbf[s & 1], tid, nth);
        }

        // layer 0 : h0(p) = tanh(x(p)@Wih0^T + h0(p-1)@Whh0^T + b0)
        if (p <= 2 * TOUT - 1) {
            const bool enc = p < TOUT;
            const unsigned short* Wih = enc ? wEWih : wDWih;
            const unsigned short* Whh = enc ? wEWhh : wDWhh;
            const float* bias = (enc ? bEnc : bDec);
            v8f acc[2] = {{0.f,0.f,0.f,0.f,0.f,0.f,0.f,0.f},
                          {0.f,0.f,0.f,0.f,0.f,0.f,0.f,0.f}};
            __syncthreads();                            // previous readers done
            stage_issue(sA0, xbf[p & 1] + (size_t)bbase * DIM, threadIdx.x);
            stage_issue(sA1, h0buf[(p + 1) & 1] + (size_t)bbase * DIM, threadIdx.x);
            stage_wait();
            mm_accum(acc, sA0, Wih + nroff, g, l15);
            mm_accum(acc, sA1, Whh + nroff, g, l15);
            const float bv = bias[ncol];
            unsigned short* dst = h0buf[p & 1];
            #pragma unroll
            for (int mt = 0; mt < 2; ++mt)
                #pragma unroll
                for (int v = 0; v < 8; ++v) {
                    const int brow = bbase + mt * 16 + v + 8 * g;
                    dst[(size_t)brow * DIM + ncol] = f2bf(tanhf(acc[mt][v] + bv));
                }
        }

        // layer 1 : h1(s) = tanh(h0(s)@Wih1^T + h1(s-1)@Whh1^T + b1), s = p-1
        if (p >= 1 && p <= 2 * TOUT) {
            const int  s   = p - 1;
            const bool enc = s < TOUT;
            const unsigned short* Wih = (enc ? wEWih : wDWih) + LINELEM;
            const unsigned short* Whh = (enc ? wEWhh : wDWhh) + LINELEM;
            const float* bias = (enc ? bEnc : bDec) + DIM;
            v8f acc[2] = {{0.f,0.f,0.f,0.f,0.f,0.f,0.f,0.f},
                          {0.f,0.f,0.f,0.f,0.f,0.f,0.f,0.f}};
            __syncthreads();
            stage_issue(sA0, h0buf[s & 1] + (size_t)bbase * DIM, threadIdx.x);
            stage_issue(sA1, h1buf[(s + 1) & 1] + (size_t)bbase * DIM, threadIdx.x);
            stage_wait();
            mm_accum(acc, sA0, Wih + nroff, g, l15);
            mm_accum(acc, sA1, Whh + nroff, g, l15);
            const float bv = bias[ncol];
            unsigned short* dst = h1buf[s & 1];
            #pragma unroll
            for (int mt = 0; mt < 2; ++mt)
                #pragma unroll
                for (int v = 0; v < 8; ++v) {
                    const int brow = bbase + mt * 16 + v + 8 * g;
                    dst[(size_t)brow * DIM + ncol] = f2bf(tanhf(acc[mt][v] + bv));
                }
        }

        // linear : logits(t) = h1(t)@linW^T + linB, t = p-2 (decoder steps only)
        if (p >= TOUT + 2) {
            const int t = p - 2;                       // 512..1023
            v8f acc[2] = {{0.f,0.f,0.f,0.f,0.f,0.f,0.f,0.f},
                          {0.f,0.f,0.f,0.f,0.f,0.f,0.f,0.f}};
            __syncthreads();
            stage_issue(sA0, h1buf[t & 1] + (size_t)bbase * DIM, threadIdx.x);
            stage_wait();
            mm_accum(acc, sA0, wLin + nroff, g, l15);
            const float bv = linB[ncol];
            float* dst = out + (size_t)(t - TOUT) * HELEM;
            #pragma unroll
            for (int mt = 0; mt < 2; ++mt)
                #pragma unroll
                for (int v = 0; v < 8; ++v) {
                    const int brow = bbase + mt * 16 + v + 8 * g;
                    dst[(size_t)brow * DIM + ncol] = acc[mt][v] + bv;
                }
        }

        if (p <= 2 * TOUT) grid_barrier(bar, epoch);
    }
}

// in-place log-softmax over rows of 1024, one wave32 per row
__global__ __launch_bounds__(NTHR) void logsoftmax_kernel(float* __restrict__ out) {
    const int row  = blockIdx.x * (NTHR / 32) + (threadIdx.x >> 5);
    const int lane = threadIdx.x & 31;
    float* p = out + (size_t)row * DIM;

    float m = -3.4e38f;
    for (int i = lane; i < DIM; i += 32) m = fmaxf(m, p[i]);
    #pragma unroll
    for (int o = 16; o > 0; o >>= 1) m = fmaxf(m, __shfl_xor(m, o, 32));

    float s = 0.f;
    for (int i = lane; i < DIM; i += 32) s += expf(p[i] - m);
    #pragma unroll
    for (int o = 16; o > 0; o >>= 1) s += __shfl_xor(s, o, 32);

    const float lse = m + logf(s);
    for (int i = lane; i < DIM; i += 32) p[i] -= lse;
}

// ---------------------------------------------------------------------------
// launch
// ---------------------------------------------------------------------------
extern "C" void kernel_launch(void* const* d_in, const int* in_sizes, int n_in,
                              void* d_out, int out_size, void* d_ws, size_t ws_size,
                              hipStream_t stream) {
    (void)in_sizes; (void)n_in; (void)out_size; (void)ws_size;
    const float* X    = (const float*)d_in[0];
    const float* Y    = (const float*)d_in[1];
    const float* eWih = (const float*)d_in[2];
    const float* eWhh = (const float*)d_in[3];
    const float* eBih = (const float*)d_in[4];
    const float* eBhh = (const float*)d_in[5];
    const float* dBih_unused = nullptr; (void)dBih_unused;
    const float* dWih = (const float*)d_in[6];
    const float* dWhh = (const float*)d_in[7];
    const float* dBih = (const float*)d_in[8];
    const float* dBhh = (const float*)d_in[9];
    const float* linW = (const float*)d_in[10];
    const float* linB = (const float*)d_in[11];

    ws_init_kernel<<<1, 32, 0, stream>>>((unsigned*)d_ws);
    rnn_persistent_kernel<<<NBLK, NTHR, SBYTES, stream>>>(
        X, Y, eWih, eWhh, eBih, eBhh, dWih, dWhh, dBih, dBhh, linW, linB,
        (float*)d_out, (char*)d_ws);
    logsoftmax_kernel<<<(TOUT * BATCH) / (NTHR / 32), NTHR, 0, stream>>>((float*)d_out);
}
